// LePEAttention_81106162418413
// MI455X (gfx1250) — matrix operand order repaired and checked by reading, not compiled, not run on hardware
//
#include <hip/hip_runtime.h>

// ---------------------------------------------------------------------------
// CSWin LePE attention (IDX=0): b=16, h=w=56, DIM=256, 8 heads, head_dim=32,
// vertical 56x7 windows -> 1024 (window, head) attention problems 392x392x32.
// Transposed flash-attention: S^T = K*Q^T, O^T = V^T*P^T via
// v_wmma_f32_16x16x32_bf16. All fragment loads are contiguous b128; softmax
// rows are per-lane (one shfl_xor(16) per reduction); P^T built in registers
// from each lane's own packed-bf16 P values. Fused depthwise-3x3 LePE epilogue.
// ---------------------------------------------------------------------------

typedef __attribute__((ext_vector_type(16))) __bf16 v16bf;
typedef __attribute__((ext_vector_type(8)))  __bf16 v8bf;
typedef __attribute__((ext_vector_type(2)))  __bf16 v2bf;
typedef __attribute__((ext_vector_type(8)))  float  v8f;
typedef __attribute__((ext_vector_type(4)))  float  v4f;

#define NUM_HEADS 8
#define HEAD_DIM  32
#define SEQ       392      // 56*7 tokens per window
#define SEQ_PAD   416      // 13 * 32
#define NKVT      13       // kv tiles of 32
#define NQT       25       // q tiles of 16 (400 >= 392)
#define HH        56
#define WW        56
#define WINW      7
#define DIM       256
#define NTOK      3136     // 56*56
#define BATCH     16
#define VT_STRIDE 424      // padded kv stride for transposed V (16B aligned)

union V16U { v16bf v; v8bf h[2]; };

__device__ __forceinline__ v8f wmma_bf16(v16bf a, v16bf b, v8f c) {
  return __builtin_amdgcn_wmma_f32_16x16x32_bf16(false, a, false, b,
                                                 (short)0, c, false, false);
}

__device__ __forceinline__ unsigned pack_bf16(float a, float b) {
  union { v2bf v; unsigned u; } x;
  x.v[0] = (__bf16)a;
  x.v[1] = (__bf16)b;
  return x.u;
}

__global__ __launch_bounds__(256, 1)
void lepe_attn_kernel(const float* __restrict__ qkv,
                      const float* __restrict__ conv_w,
                      const float* __restrict__ conv_b,
                      float* __restrict__ out)
{
  __shared__ __align__(16) __bf16 Kl[SEQ_PAD * HEAD_DIM];     // 26.0 KB row-major
  __shared__ __align__(16) __bf16 Vt[HEAD_DIM * VT_STRIDE];   // 26.5 KB transposed
  __shared__ __align__(16) float  Vf[SEQ * HEAD_DIM];         // 49.0 KB fp32 V (LePE)
  __shared__ __align__(16) float  Wt[9 * HEAD_DIM];           //  1.1 KB conv w (this head)

  const int bid    = blockIdx.x;       // 0..1023
  const int head   = bid & 7;
  const int win    = bid >> 3;         // 0..127
  const int b_idx  = win >> 3;
  const int win_x  = win & 7;

  const int tid    = threadIdx.x;
  const int lane   = tid & 31;
  const int wid    = tid >> 5;         // wave id 0..7
  const int laneHi = (lane >> 4) & 1;
  const int lane15 = lane & 15;
  const int off    = laneHi * 8;       // A-fragment K-chunk base

  const float scale = 0.17677669529663687f;   // 32^-0.5

  const float* qp = qkv;
  const float* kp = qkv + (size_t)1 * BATCH * NTOK * DIM;
  const float* vp = qkv + (size_t)2 * BATCH * NTOK * DIM;

  // -------- cooperative stage: K (row-major bf16), V^T (bf16), V (fp32) ----
  for (int e = tid; e < SEQ_PAD * HEAD_DIM; e += 256) {
    int t = e >> 5, d = e & 31;
    float kv = 0.f, vv = 0.f;
    if (t < SEQ) {
      int nidx = (t / WINW) * WW + win_x * WINW + (t % WINW);
      size_t g = ((size_t)b_idx * NTOK + nidx) * DIM + head * HEAD_DIM + d;
      kv = kp[g]; vv = vp[g];
      Vf[e] = vv;
    }
    Kl[e] = (__bf16)kv;
    Vt[d * VT_STRIDE + t] = (__bf16)vv;
  }
  // depthwise conv weights for this head's 32 channels, transposed [tap][d]
  for (int e = tid; e < 9 * HEAD_DIM; e += 256) {
    int tap = e >> 5, d = e & 31;
    Wt[tap * HEAD_DIM + d] = conv_w[(head * HEAD_DIM + d) * 9 + tap];
  }
  __syncthreads();

  // each wave owns q tiles wid, wid+8, wid+16, (wid+24)
  for (int qt = wid; qt < NQT; qt += 8) {
    const int qbase = qt * 16;

    // ---- B fragment of Q^T: lane = q col, elems = dims 8*laneHi+0..7, 16+8*laneHi+0..7 ----
    V16U bq;
    {
      int qrow = qbase + lane15;
      if (qrow < SEQ) {
        int nidx = (qrow / WINW) * WW + win_x * WINW + (qrow % WINW);
        const float* qrp = qp + ((size_t)b_idx * NTOK + nidx) * DIM
                              + head * HEAD_DIM + laneHi * 8;
        v4f f0 = *(const v4f*)(qrp);
        v4f f1 = *(const v4f*)(qrp + 4);
        v4f f2 = *(const v4f*)(qrp + 16);
        v4f f3 = *(const v4f*)(qrp + 20);
        #pragma unroll
        for (int j = 0; j < 4; ++j) {
          bq.v[j]      = (__bf16)(f0[j] * scale);
          bq.v[4 + j]  = (__bf16)(f1[j] * scale);
          bq.v[8 + j]  = (__bf16)(f2[j] * scale);
          bq.v[12 + j] = (__bf16)(f3[j] * scale);
        }
      } else {
        #pragma unroll
        for (int j = 0; j < 16; ++j) bq.v[j] = (__bf16)0.f;
      }
    }

    v8f ot0 = {};  // O^T dims 0..15  (lane: q=L%16, row r -> dim 8*laneHi+r)
    v8f ot1 = {};  // O^T dims 16..31
    float mrun = -1e30f, lrun = 0.f;   // per-lane (per-query) softmax state

    for (int kt = 0; kt < NKVT; ++kt) {
      const int kvbase = kt * 32;

      // ---- A fragments of K: rows = kv, two contiguous b128 chunks ----
      V16U ak_lo, ak_hi;
      {
        const __bf16* r0 = Kl + (kvbase + lane15) * HEAD_DIM + off;
        const __bf16* r1 = Kl + (kvbase + 16 + lane15) * HEAD_DIM + off;
        ak_lo.h[0] = *(const v8bf*)(r0);
        ak_lo.h[1] = *(const v8bf*)(r0 + 16);
        ak_hi.h[0] = *(const v8bf*)(r1);
        ak_hi.h[1] = *(const v8bf*)(r1 + 16);
      }

      v8f zc = {};
      v8f stl = wmma_bf16(ak_lo.v, bq.v, zc);  // S^T kv [kvbase..+15] x q
      v8f sth = wmma_bf16(ak_hi.v, bq.v, zc);  // S^T kv [kvbase+16..+31] x q

      // mask padded kv rows (only last tile)
      if (kvbase + 32 > SEQ) {
        #pragma unroll
        for (int r = 0; r < 8; ++r) {
          if (kvbase + off + r >= SEQ)      stl[r] = -1e30f;
          if (kvbase + 16 + off + r >= SEQ) sth[r] = -1e30f;
        }
      }

      // ---- online softmax: full row per lane-pair (L, L^16) ----
      float mx = stl[0];
      #pragma unroll
      for (int r = 1; r < 8; ++r) mx = fmaxf(mx, stl[r]);
      #pragma unroll
      for (int r = 0; r < 8; ++r) mx = fmaxf(mx, sth[r]);
      mx = fmaxf(mx, __shfl_xor(mx, 16, 32));
      float mnew = fmaxf(mrun, mx);

      float p_lo[8], p_hi[8], sum = 0.f;
      #pragma unroll
      for (int r = 0; r < 8; ++r) { p_lo[r] = __expf(stl[r] - mnew); sum += p_lo[r]; }
      #pragma unroll
      for (int r = 0; r < 8; ++r) { p_hi[r] = __expf(sth[r] - mnew); sum += p_hi[r]; }
      sum += __shfl_xor(sum, 16, 32);

      float sc = __expf(mrun - mnew);
      lrun = lrun * sc + sum;
      mrun = mnew;
      #pragma unroll
      for (int r = 0; r < 8; ++r) { ot0[r] *= sc; ot1[r] *= sc; }

      // ---- build P^T B-fragment in registers (packed bf16, own lane's values) ----
      // lane L holds q col L%16; needs kv 8*laneHi..+7, 16+8*laneHi..+7 (already held).
      unsigned ulo[4], uhi[4];
      #pragma unroll
      for (int i = 0; i < 4; ++i) {
        ulo[i] = pack_bf16(p_lo[2 * i], p_lo[2 * i + 1]); // own kv  8*laneHi+2i..
        uhi[i] = pack_bf16(p_hi[2 * i], p_hi[2 * i + 1]); // own kv 16+8*laneHi+2i..
      }
      unsigned olo[4], ohi[4];
      #pragma unroll
      for (int i = 0; i < 4; ++i) {
        olo[i] = __shfl_xor(ulo[i], 16, 32);
        ohi[i] = __shfl_xor(uhi[i], 16, 32);
      }
      union { v16bf v; unsigned u[8]; } bp;
      #pragma unroll
      for (int i = 0; i < 4; ++i) {
        bp.u[i]     = ulo[i];  // kv 8*laneHi + 2i..
        bp.u[4 + i] = uhi[i];  // kv 16 + 8*laneHi + 2i..
      }

      // ---- A fragments of V^T: rows = dim, contiguous kv chunks ----
      V16U av0, av1;
      {
        const __bf16* v0 = Vt + lane15 * VT_STRIDE + kvbase + off;
        const __bf16* v1 = Vt + (16 + lane15) * VT_STRIDE + kvbase + off;
        av0.h[0] = *(const v8bf*)(v0);
        av0.h[1] = *(const v8bf*)(v0 + 16);
        av1.h[0] = *(const v8bf*)(v1);
        av1.h[1] = *(const v8bf*)(v1 + 16);
      }

      ot0 = wmma_bf16(av0.v, bp.v, ot0);
      ot1 = wmma_bf16(av1.v, bp.v, ot1);
    }

    // ---- epilogue: 1/l, fused 3x3 depthwise LePE, vectorized store ----
    int t = qbase + lane15;
    if (t < SEQ) {
      float rl = (lrun > 0.f) ? (1.f / lrun) : 0.f;
      int ti = t / WINW, tj = t % WINW;
      int db0 = laneHi * 8;        // ot0 covers dims db0..db0+7
      int db1 = 16 + laneHi * 8;   // ot1 covers dims db1..db1+7

      const float* cb = conv_b + head * HEAD_DIM;
      v4f a00 = *(const v4f*)(cb + db0);
      v4f a01 = *(const v4f*)(cb + db0 + 4);
      v4f a10 = *(const v4f*)(cb + db1);
      v4f a11 = *(const v4f*)(cb + db1 + 4);

      #pragma unroll
      for (int di = 0; di < 3; ++di) {
        int ii = ti + di - 1;
        if (ii < 0 || ii >= HH) continue;
        #pragma unroll
        for (int dj = 0; dj < 3; ++dj) {
          int jj = tj + dj - 1;
          if (jj < 0 || jj >= WINW) continue;
          const float* wt = Wt + (di * 3 + dj) * HEAD_DIM;
          const float* vf = Vf + (ii * WINW + jj) * HEAD_DIM;
          a00 += (*(const v4f*)(wt + db0)) * (*(const v4f*)(vf + db0));
          a01 += (*(const v4f*)(wt + db0 + 4)) * (*(const v4f*)(vf + db0 + 4));
          a10 += (*(const v4f*)(wt + db1)) * (*(const v4f*)(vf + db1));
          a11 += (*(const v4f*)(wt + db1 + 4)) * (*(const v4f*)(vf + db1 + 4));
        }
      }

      int nidx = ti * WW + win_x * WINW + tj;
      float* op = out + ((size_t)b_idx * NTOK + nidx) * DIM + head * HEAD_DIM;
      v4f s00, s01, s10, s11;
      #pragma unroll
      for (int j = 0; j < 4; ++j) {
        s00[j] = ot0[j] * rl + a00[j];
        s01[j] = ot0[4 + j] * rl + a01[j];
        s10[j] = ot1[j] * rl + a10[j];
        s11[j] = ot1[4 + j] * rl + a11[j];
      }
      *(v4f*)(op + db0)     = s00;
      *(v4f*)(op + db0 + 4) = s01;
      *(v4f*)(op + db1)     = s10;
      *(v4f*)(op + db1 + 4) = s11;
    }
  }
}

extern "C" void kernel_launch(void* const* d_in, const int* in_sizes, int n_in,
                              void* d_out, int out_size, void* d_ws, size_t ws_size,
                              hipStream_t stream) {
  (void)in_sizes; (void)n_in; (void)out_size; (void)d_ws; (void)ws_size;
  const float* qkv    = (const float*)d_in[0];
  const float* conv_w = (const float*)d_in[1];
  const float* conv_b = (const float*)d_in[2];
  float* out          = (float*)d_out;

  dim3 grid(128 * NUM_HEADS);   // 1024 (window, head) problems
  dim3 block(256);              // 8 waves of 32
  lepe_attn_kernel<<<grid, block, 0, stream>>>(qkv, conv_w, conv_b, out);
}
